// conv_caps_DR_23596550324376
// MI455X (gfx1250) — matrix operand order repaired and checked by reading, not compiled, not run on hardware
//
#include <hip/hip_runtime.h>
#include <hip/hip_bf16.h>

// ---------------- problem constants ----------------
#define BB    8
#define HH    12
#define WW    12
#define NX    (BB*HH*WW)     // 1152 pixels
#define ICAPS 32
#define KK    3
#define IC    (KK*KK*ICAPS)  // 288 input "rows"
#define IP    16             // input pose
#define OC    32             // output caps
#define OP    16             // output pose
#define ITERS 3
#define NBLK  (NX/16)        // 72 blocks of 16 pixels

typedef __attribute__((ext_vector_type(16))) __bf16 v16bf;
typedef __attribute__((ext_vector_type(8)))  __bf16 v8bf;
typedef __attribute__((ext_vector_type(8)))  float  v8f;

// ---------------- prep kernels ----------------

// im2col (SAME pad, stride 1) + fp32 -> bf16.  u[n][i][p]
__global__ void prep_u_kernel(const float* __restrict__ x, __bf16* __restrict__ u) {
    int idx = blockIdx.x * blockDim.x + threadIdx.x;
    if (idx >= NX * IC * IP) return;
    int p = idx % IP;
    int i = (idx / IP) % IC;
    int n = idx / (IP * IC);
    int b = n / (HH * WW);
    int y = (n / WW) % HH;
    int xx = n % WW;
    int kk = i / ICAPS;
    int icap = i % ICAPS;
    int yy = y + kk / KK - 1;
    int xw = xx + kk % KK - 1;
    float v = 0.0f;
    if (yy >= 0 && yy < HH && xw >= 0 && xw < WW)
        v = x[((((size_t)b * HH + yy) * WW + xw) * ICAPS + icap) * IP + p];
    u[idx] = (__bf16)v;
}

// W fp32 [i][c][p][q]  ->  Wq bf16 [i][c][p][q]  and  Wt bf16 [i][c][q][p]
__global__ void prep_w_kernel(const float* __restrict__ Wf,
                              __bf16* __restrict__ Wq, __bf16* __restrict__ Wt) {
    int idx = blockIdx.x * blockDim.x + threadIdx.x;
    if (idx >= IC * OC * IP * OP) return;
    int q = idx % OP;
    int p = (idx / OP) % IP;
    int c = (idx / (OP * IP)) % OC;
    int i = idx / (OP * IP * OC);
    float v = Wf[idx];
    Wq[idx] = (__bf16)v;
    Wt[((((size_t)i * OC + c) * OP + q) * IP) + p] = (__bf16)v;
}

__global__ void fill_kernel(float* __restrict__ ptr, float val, int count) {
    int idx = blockIdx.x * blockDim.x + threadIdx.x;
    if (idx < count) ptr[idx] = val;
}

// softmax over the 32 output caps.
// logits / cc layout: [c][i][n]  (n fastest -> coalesced across lanes everywhere)
__global__ void softmax_kernel(const float* __restrict__ logits, float* __restrict__ cc) {
    int t = blockIdx.x * blockDim.x + threadIdx.x;
    if (t >= NX * IC) return;
    int n = t % NX;
    int i = t / NX;
    size_t base = (size_t)i * NX + n;            // + c*IC*NX per capsule
    const size_t cstride = (size_t)IC * NX;
    float e[OC];
    float m = -3.0e38f;
    #pragma unroll
    for (int c = 0; c < OC; ++c) { e[c] = logits[base + (size_t)c * cstride]; m = fmaxf(m, e[c]); }
    float sum = 0.0f;
    #pragma unroll
    for (int c = 0; c < OC; ++c) { e[c] = __expf(e[c] - m); sum += e[c]; }
    float inv = 1.0f / sum;
    #pragma unroll
    for (int c = 0; c < OC; ++c) cc[base + (size_t)c * cstride] = e[c] * inv;
}

// ---------------- WMMA s-kernel ----------------
// s[n,c,q] = sum_{i,p} (cc[c,i,n]*u[n,i,p]) * W[i,c,p,q]
// grid (NBLK, OC/4); 4 waves/block, wave w handles capsule c = by*4+w.
// M = 16 pixels, N = 16 pose dims, K = 32 = two (i, p=0..15) blocks.
// A-pack uses packed bf16 multiplies (v_pk_mul_bf16): v8bf LDS load * splat(cc).
__global__ void s_wmma_kernel(const __bf16* __restrict__ u,   // [NX][IC][IP]
                              const __bf16* __restrict__ Wt,  // [IC][OC][OP(q)][IP(p)]
                              const float* __restrict__ cc,   // [OC][IC][NX]
                              float* __restrict__ s_out) {    // [NX][OC][OP]
    extern __shared__ __bf16 uls[];                 // [16][IC][IP] = 144 KB
    const int nb   = blockIdx.x;
    const int wid  = threadIdx.x >> 5;
    const int lane = threadIdx.x & 31;
    const int c    = blockIdx.y * 4 + wid;

    // stage the 16-pixel u block in LDS (shared by the 4 waves)
    {
        const uint32_t* src = (const uint32_t*)(u + (size_t)nb * 16 * IC * IP);
        uint32_t* dst = (uint32_t*)uls;
        for (int t = threadIdx.x; t < 16 * IC * IP / 2; t += blockDim.x) dst[t] = src[t];
    }
    __syncthreads();

    const int nloc = lane & 15;      // A-row M / D column N
    const int hsel = lane >> 4;      // lane half
    const int n    = nb * 16 + nloc;
    const float* ccb = cc + (size_t)c * IC * NX + n;   // + i*NX per input row

    v8f acc0 = {};
    v8f acc1 = {};
    for (int it = 0; it < IC / 4; ++it) {
        const int i0 = it * 4;
        // ---- k-step A: rows i0, i0+1 -> acc0 ----
        {
            const __bf16 b0 = (__bf16)ccb[(size_t)(i0    ) * NX];
            const __bf16 b1 = (__bf16)ccb[(size_t)(i0 + 1) * NX];
            const __bf16* urow0 = &uls[((size_t)nloc * IC + i0) * IP + hsel * 8];
            // lanes<16 carry K = {0..7, 16..23} (p=0..7); lanes>=16 K = {8..15, 24..31}
            v8bf a0 = *(const v8bf*)(urow0)      * b0;   // row i0
            v8bf a1 = *(const v8bf*)(urow0 + IP) * b1;   // row i0+1
            v16bf A = __builtin_shufflevector(a0, a1, 0, 1, 2, 3, 4, 5, 6, 7,
                                                      8, 9, 10, 11, 12, 13, 14, 15);
            const __bf16* wrow = &Wt[((((size_t)(i0)) * OC + c) * OP + nloc) * IP + hsel * 8];
            v16bf Bv = __builtin_shufflevector(*(const v8bf*)wrow, *(const v8bf*)(wrow + OC * OP * IP),
                                               0, 1, 2, 3, 4, 5, 6, 7, 8, 9, 10, 11, 12, 13, 14, 15);
            acc0 = __builtin_amdgcn_wmma_f32_16x16x32_bf16(false, A, false, Bv,
                                                           (short)0, acc0, false, false);
        }
        // ---- k-step B: rows i0+2, i0+3 -> acc1 ----
        {
            const __bf16 b0 = (__bf16)ccb[(size_t)(i0 + 2) * NX];
            const __bf16 b1 = (__bf16)ccb[(size_t)(i0 + 3) * NX];
            const __bf16* urow0 = &uls[((size_t)nloc * IC + i0 + 2) * IP + hsel * 8];
            v8bf a0 = *(const v8bf*)(urow0)      * b0;
            v8bf a1 = *(const v8bf*)(urow0 + IP) * b1;
            v16bf A = __builtin_shufflevector(a0, a1, 0, 1, 2, 3, 4, 5, 6, 7,
                                                      8, 9, 10, 11, 12, 13, 14, 15);
            const __bf16* wrow = &Wt[((((size_t)(i0 + 2)) * OC + c) * OP + nloc) * IP + hsel * 8];
            v16bf Bv = __builtin_shufflevector(*(const v8bf*)wrow, *(const v8bf*)(wrow + OC * OP * IP),
                                               0, 1, 2, 3, 4, 5, 6, 7, 8, 9, 10, 11, 12, 13, 14, 15);
            acc1 = __builtin_amdgcn_wmma_f32_16x16x32_bf16(false, A, false, Bv,
                                                           (short)0, acc1, false, false);
        }
    }
    // D: lane holds column q=nloc, rows m = r + 8*hsel
    #pragma unroll
    for (int r = 0; r < 8; ++r) {
        int m = r + 8 * hsel;
        s_out[(((size_t)(nb * 16 + m)) * OC + c) * OP + nloc] = acc0[r] + acc1[r];
    }
}

// ---------------- bias + squash ----------------
__global__ void squash_kernel(const float* __restrict__ s, const float* __restrict__ biases,
                              float* __restrict__ vout) {
    int idx = blockIdx.x * blockDim.x + threadIdx.x;   // (n, c)
    if (idx >= NX * OC) return;
    int c = idx % OC;
    const float* sr = s + (size_t)idx * OP;
    const float* br = biases + c * OP;
    float t[OP]; float sq = 0.0f;
    #pragma unroll
    for (int q = 0; q < OP; ++q) { t[q] = sr[q] + br[q]; sq += t[q] * t[q]; }
    float scale = (sq / (1.0f + sq)) * rsqrtf(sq + 1e-9f);
    float* out = vout + (size_t)idx * OP;
    #pragma unroll
    for (int q = 0; q < OP; ++q) out[q] = t[q] * scale;
}

// ---------------- WMMA agreement-update ----------------
// logits[c,i,n] += sum_p u[n,i,p] * Wv[p,n],  Wv[p,n] = sum_q W[i,c,p,q]*v[n,c,q]
// grid (NBLK, OC); 4 waves/block loop over i.  WMMA: M=p, K=q (zero-padded to 32), N=n.
__global__ void update_wmma_kernel(const __bf16* __restrict__ u,   // [NX][IC][IP]
                                   const __bf16* __restrict__ Wq,  // [IC][OC][IP(p)][OP(q)]
                                   const float* __restrict__ vbuf, // [NX][OC][OP]
                                   float* __restrict__ logits) {   // [OC][IC][NX]
    extern __shared__ __bf16 uls[];
    const int nb   = blockIdx.x;
    const int c    = blockIdx.y;
    const int wid  = threadIdx.x >> 5;
    const int lane = threadIdx.x & 31;
    {
        const uint32_t* src = (const uint32_t*)(u + (size_t)nb * 16 * IC * IP);
        uint32_t* dst = (uint32_t*)uls;
        for (int t = threadIdx.x; t < 16 * IC * IP / 2; t += blockDim.x) dst[t] = src[t];
    }
    __syncthreads();

    const int nloc = lane & 15;
    const int hsel = lane >> 4;
    const v8bf zv = {};

    // B tile (constant across i): lanes<16 carry K=q=0..7, lanes>=16 K=q=8..15; elements 8..15 (K>=16) are zero pad
    v16bf Bv;
    {
        const float* vr = &vbuf[(((size_t)(nb * 16 + nloc)) * OC + c) * OP + hsel * 8];
        #pragma unroll
        for (int e = 0; e < 8; ++e) Bv[e] = (__bf16)vr[e];
        #pragma unroll
        for (int e = 8; e < 16; ++e) Bv[e] = (__bf16)0.0f;
    }

    float* lgb = logits + (size_t)c * IC * NX + nb * 16 + nloc;    // + i*NX per row

    for (int i = wid; i < IC; i += 4) {
        // A tile: M=p=nloc; lanes<16: K=q=0..7 live, lanes>=16: K=q=8..15 live; K>=16 zero
        const __bf16* wrow = &Wq[((((size_t)i) * OC + c) * IP + nloc) * OP + hsel * 8];
        v8bf wv = *(const v8bf*)wrow;                    // one 16B load
        v16bf A = __builtin_shufflevector(wv, zv, 0, 1, 2, 3, 4, 5, 6, 7,
                                                  8, 9, 10, 11, 12, 13, 14, 15);
        v8f acc = {};
        acc = __builtin_amdgcn_wmma_f32_16x16x32_bf16(false, A, false, Bv,
                                                      (short)0, acc, false, false);
        // D: lane column n=nloc, rows p = r + 8*hsel;  dot with u over p, fold halves
        float part = 0.0f;
        #pragma unroll
        for (int r = 0; r < 8; ++r) {
            int p = r + 8 * hsel;
            part += acc[r] * (float)uls[((size_t)nloc * IC + i) * IP + p];
        }
        part += __shfl_xor(part, 16, 32);
        if (hsel == 0) {
            lgb[(size_t)i * NX] += part;          // coalesced across lanes (n contiguous)
        }
    }
}

// ---------------- host launcher ----------------
extern "C" void kernel_launch(void* const* d_in, const int* in_sizes, int n_in,
                              void* d_out, int out_size, void* d_ws, size_t ws_size,
                              hipStream_t stream) {
    const float* x  = (const float*)d_in[0];   // (8,12,12,32,16)
    const float* Wf = (const float*)d_in[1];   // (288,32,16,16)
    const float* bi = (const float*)d_in[2];   // (32,16)
    float* out = (float*)d_out;                // (8,12,12,32,16) == [NX][OC][OP]

    // carve workspace (256B aligned)
    char* ws = (char*)d_ws;
    size_t off = 0;
    #define CARVE(name, bytes) void* name = (void*)(ws + off); off += (((size_t)(bytes)) + 255) & ~(size_t)255;
    CARVE(u_bf,   (size_t)NX * IC * IP * sizeof(__bf16));          // 10.6 MB
    CARVE(Wq_bf,  (size_t)IC * OC * IP * OP * sizeof(__bf16));     // 4.7 MB
    CARVE(Wt_bf,  (size_t)IC * OC * IP * OP * sizeof(__bf16));     // 4.7 MB
    CARVE(logits, (size_t)OC * IC * NX * sizeof(float));           // 42.5 MB  [c][i][n]
    CARVE(ccbuf,  (size_t)OC * IC * NX * sizeof(float));           // 42.5 MB  [c][i][n]
    CARVE(sbuf,   (size_t)NX * OC * OP * sizeof(float));           // 2.4 MB
    CARVE(vbuf,   (size_t)NX * OC * OP * sizeof(float));           // 2.4 MB
    #undef CARVE
    (void)ws_size; (void)in_sizes; (void)n_in; (void)out_size;

    const size_t ldsBytes = (size_t)16 * IC * IP * sizeof(__bf16); // 147456 B

    // prep
    {
        int tot = NX * IC * IP;
        prep_u_kernel<<<(tot + 255) / 256, 256, 0, stream>>>(x, (__bf16*)u_bf);
    }
    {
        int tot = IC * OC * IP * OP;
        prep_w_kernel<<<(tot + 255) / 256, 256, 0, stream>>>(Wf, (__bf16*)Wq_bf, (__bf16*)Wt_bf);
    }
    {
        int tot = OC * IC * NX;
        fill_kernel<<<(tot + 255) / 256, 256, 0, stream>>>((float*)logits, 0.0f, tot);
    }

    // dynamic routing
    for (int ii = 0; ii < ITERS; ++ii) {
        if (ii == 0) {
            // softmax of all-zero logits is uniform 1/OC: skip the 42MB pass
            int tot = OC * IC * NX;
            fill_kernel<<<(tot + 255) / 256, 256, 0, stream>>>((float*)ccbuf,
                                                               1.0f / (float)OC, tot);
        } else {
            int rows = NX * IC;
            softmax_kernel<<<(rows + 255) / 256, 256, 0, stream>>>((const float*)logits,
                                                                   (float*)ccbuf);
        }
        s_wmma_kernel<<<dim3(NBLK, OC / 4), 128, ldsBytes, stream>>>(
            (const __bf16*)u_bf, (const __bf16*)Wt_bf, (const float*)ccbuf, (float*)sbuf);

        float* vdst = (ii == ITERS - 1) ? out : (float*)vbuf;
        {
            int tot = NX * OC;
            squash_kernel<<<(tot + 255) / 256, 256, 0, stream>>>((const float*)sbuf, bi, vdst);
        }
        if (ii < ITERS - 1) {
            update_wmma_kernel<<<dim3(NBLK, OC), 128, ldsBytes, stream>>>(
                (const __bf16*)u_bf, (const __bf16*)Wq_bf, (const float*)vbuf, (float*)logits);
        }
    }
}
